// RDB_Horizontal_Conv_3D_77979426226523
// MI455X (gfx1250) — matrix operand order repaired and checked by reading, not compiled, not run on hardware
//
#include <hip/hip_runtime.h>

// ---------------------------------------------------------------------------
// Locally-connected 3D conv (per-z weights) for MI455X / gfx1250.
// Path: f32 -> f16 staging (L2-resident), v_wmma_f32_16x16x32_f16 main loop,
// f32 accumulate, leaky-ReLU, concat([x, out], axis=1).
//
// Shapes: x [4,64,128,128,10], W [10,32,64,3,3,3], out [4,96,128,128,10].
// Workspace: xs f16 (83,886,080 B) + wA f16 (1,105,920 B) ~= 85 MB in d_ws.
// ---------------------------------------------------------------------------

typedef _Float16 v8h  __attribute__((ext_vector_type(8)));
typedef _Float16 v16h __attribute__((ext_vector_type(16)));
typedef float    v8f  __attribute__((ext_vector_type(8)));

#define NEG_SLOPE 0.2f

constexpr int B_ = 4, C_ = 64, D_ = 128, H_ = 128, Z_ = 10, O_ = 32;
constexpr size_t XS_BYTES = (size_t)B_ * D_ * H_ * Z_ * C_ * 2;   // 83,886,080

__device__ __forceinline__ v16h cat16(v8h lo, v8h hi) {
  return __builtin_shufflevector(lo, hi, 0,1,2,3,4,5,6,7,8,9,10,11,12,13,14,15);
}

// x [B,C,D,H,Z] f32  ->  xs [B,D,H,Z,C] f16  (c contiguous: B-fragment friendly)
__global__ __launch_bounds__(256) void stage_x(const float* __restrict__ x,
                                               _Float16* __restrict__ xs) {
  size_t tid = (size_t)blockIdx.x * 256 + threadIdx.x;   // 41,943,040 total
  int c = tid & 63;  size_t r = tid >> 6;
  int z = (int)(r % 10); r /= 10;
  int h = (int)(r & 127); r >>= 7;
  int d = (int)(r & 127); int b = (int)(r >> 7);
  float v = x[((((size_t)b * C_ + c) * D_ + d) * H_ + h) * Z_ + z];
  xs[tid] = (_Float16)v;
}

// W [Z,O,C,3,3,3] f32 -> wA [z][tap][q][m][lane][16] f16  (ISA A 16x32 layout)
__global__ __launch_bounds__(256) void stage_w(const float* __restrict__ w,
                                               _Float16* __restrict__ wA) {
  int tid = blockIdx.x * 256 + threadIdx.x;              // 552,960 total
  int e    = tid & 15;
  int lane = (tid >> 4) & 31;
  int r    = tid >> 9;
  int m = r & 1;  r >>= 1;
  int q = r & 1;  r >>= 1;
  int t = r % 27; int z = r / 27;
  int kd = t / 9, kh = (t / 3) % 3, kw = t % 3;
  // 16-bit A-matrix 16x32 per-lane K mapping (ISA 7.12.2):
  int k = (lane < 16) ? ((e < 8) ? e : e + 8)
                      : ((e < 8) ? e + 8 : e + 16);
  int c = q * 32 + k;
  int o = m * 16 + (lane & 15);
  float v = w[((((((size_t)z * O_ + o) * C_ + c) * 3 + kd) * 3 + kh) * 3 + kw)];
  wA[tid] = (_Float16)v;
}

// concat part 1: out[:, 0:64] = x   (float4 streaming copy)
__global__ __launch_bounds__(256) void copy_x(const float4* __restrict__ x4,
                                              float4* __restrict__ out4) {
  size_t n = (size_t)blockIdx.x * 256 + threadIdx.x;     // 10,485,760 total
  const size_t per_b_in  = (size_t)C_ * D_ * H_ * Z_ / 4;   // 2,621,440
  const size_t per_b_out = (size_t)96 * D_ * H_ * Z_ / 4;   // 3,932,160
  size_t b = n / per_b_in, rem = n % per_b_in;
  out4[b * per_b_out + rem] = x4[n];
}

// Main GEMM: per wave a 32(O) x 32(h) tile at fixed (b,z,d). K = 27 taps x 64 ch.
__global__ __launch_bounds__(256) void conv_wmma(const _Float16* __restrict__ xs,
                                                 const _Float16* __restrict__ wA,
                                                 float* __restrict__ out) {
  const int lane  = threadIdx.x & 31;
  const int wave  = threadIdx.x >> 5;
  int blk = blockIdx.x;                                  // 2560 blocks
  const int hSeg = blk & 3;  blk >>= 2;
  const int dBlk = blk & 15; blk >>= 4;
  const int z = blk % 10;
  const int b = blk / 10;
  const int d  = dBlk * 8 + wave;
  const int h0 = hSeg * 32;
  const int col   = lane & 15;
  const int khalf = lane >> 4;

  v8f acc00 = {}, acc01 = {}, acc10 = {}, acc11 = {};
  const v8h zero8 = {};

#pragma unroll
  for (int t = 0; t < 27; ++t) {
    const int kd = t / 9, kh = (t / 3) % 3, kw = t % 3;
    const int dd = d + kd - 1;
    const int zz = z + kw - 1;
    if (dd < 0 || dd >= D_ || zz < 0 || zz >= Z_) continue;  // wave-uniform skip

    const int hh0 = h0 + col + kh - 1;     // fragment-0 pixel
    const int hh1 = hh0 + 16;              // fragment-1 pixel
    const bool ok0 = (hh0 >= 0) & (hh0 < H_);
    const bool ok1 = (hh1 >= 0) & (hh1 < H_);
    const _Float16* px0 =
        xs + ((((size_t)b * D_ + dd) * H_ + (ok0 ? hh0 : 0)) * Z_ + zz) * C_;
    const _Float16* px1 =
        xs + ((((size_t)b * D_ + dd) * H_ + (ok1 ? hh1 : 0)) * Z_ + zz) * C_;
    const _Float16* wt = wA + ((size_t)(z * 27 + t)) * 2048 + (size_t)lane * 16;

#pragma unroll
    for (int q = 0; q < 2; ++q) {          // two 32-channel K chunks
      const _Float16* wq = wt + q * 1024;
      v16h a0 = cat16(*(const v8h*)(wq      ), *(const v8h*)(wq + 8  ));  // o 0..15
      v16h a1 = cat16(*(const v8h*)(wq + 512), *(const v8h*)(wq + 520));  // o 16..31

      const int coff = q * 32 + khalf * 8;    // B layout: lane>=16 holds K=8..15,24..31
      v8h b0lo = zero8, b0hi = zero8, b1lo = zero8, b1hi = zero8;
      if (ok0) { b0lo = *(const v8h*)(px0 + coff); b0hi = *(const v8h*)(px0 + coff + 16); }
      if (ok1) { b1lo = *(const v8h*)(px1 + coff); b1hi = *(const v8h*)(px1 + coff + 16); }
      v16h bb0 = cat16(b0lo, b0hi);
      v16h bb1 = cat16(b1lo, b1hi);

      acc00 = __builtin_amdgcn_wmma_f32_16x16x32_f16(false, a0, false, bb0,
                                                     (short)0, acc00, false, false);
      acc01 = __builtin_amdgcn_wmma_f32_16x16x32_f16(false, a0, false, bb1,
                                                     (short)0, acc01, false, false);
      acc10 = __builtin_amdgcn_wmma_f32_16x16x32_f16(false, a1, false, bb0,
                                                     (short)0, acc10, false, false);
      acc11 = __builtin_amdgcn_wmma_f32_16x16x32_f16(false, a1, false, bb1,
                                                     (short)0, acc11, false, false);
    }
  }

  // Epilogue: leaky-ReLU + scatter per C/D layout (VGPR r: M = r + 8*khalf, N = col)
  const size_t DHZ = (size_t)D_ * H_ * Z_;
  float* ob = out + (size_t)b * 96 * DHZ + ((size_t)d * H_ + h0 + col) * Z_ + z;
#pragma unroll
  for (int m = 0; m < 2; ++m) {
#pragma unroll
    for (int j = 0; j < 2; ++j) {
      v8f a = (m == 0) ? (j == 0 ? acc00 : acc01) : (j == 0 ? acc10 : acc11);
#pragma unroll
      for (int r = 0; r < 8; ++r) {
        int o = 64 + m * 16 + khalf * 8 + r;
        float v = a[r];
        v = (v >= 0.0f) ? v : NEG_SLOPE * v;
        ob[(size_t)o * DHZ + (size_t)j * 16 * Z_] = v;
      }
    }
  }
}

extern "C" void kernel_launch(void* const* d_in, const int* in_sizes, int n_in,
                              void* d_out, int out_size, void* d_ws, size_t ws_size,
                              hipStream_t stream) {
  const float* x = (const float*)d_in[0];
  const float* w = (const float*)d_in[1];
  float* out = (float*)d_out;

  _Float16* xs = (_Float16*)d_ws;
  _Float16* wA = (_Float16*)((char*)d_ws + XS_BYTES);   // 256B-aligned offset

  hipLaunchKernelGGL(stage_x, dim3(163840), dim3(256), 0, stream, x, xs);
  hipLaunchKernelGGL(stage_w, dim3(2160),   dim3(256), 0, stream, w, wA);
  hipLaunchKernelGGL(copy_x,  dim3(40960),  dim3(256), 0, stream,
                     (const float4*)x, (float4*)out);
  hipLaunchKernelGGL(conv_wmma, dim3(2560), dim3(256), 0, stream, xs, wA, out);
}
